// JointERE_17952963298106
// MI455X (gfx1250) — compile-verified, matrix operand order checked by reading, not run on hardware
//
#include <hip/hip_runtime.h>
#include <hip/hip_bf16.h>

typedef __attribute__((ext_vector_type(16))) _Float16 v16h;
typedef __attribute__((ext_vector_type(8)))  _Float16 v8h;
typedef __attribute__((ext_vector_type(8)))  float    v8f;

// ---------------------------------------------------------------------------
// CDNA5 async global->LDS helpers (ASYNCcnt path)
// ---------------------------------------------------------------------------
#if defined(__has_builtin)
# if __has_builtin(__builtin_amdgcn_global_load_async_to_lds_b128)
#  define ASYNC_VIA_BUILTIN 1
# endif
# if __has_builtin(__builtin_amdgcn_s_wait_asynccnt)
#  define WAIT_ASYNC(n) __builtin_amdgcn_s_wait_asynccnt(n)
# endif
#endif
#ifndef WAIT_ASYNC
# define WAIT_ASYNC(n) asm volatile("s_wait_asynccnt %0" :: "i"(n) : "memory")
#endif

// Builtin prototype is (v4i32 AS1*, v4i32 AS3*, imm int, imm int) with
// gcc-style vector_size vectors (per hipcc diagnostic).
typedef int v4i_vec __attribute__((vector_size(16)));
typedef __attribute__((address_space(1))) v4i_vec* gptr_v4i;   // global
typedef __attribute__((address_space(3))) v4i_vec* lptr_v4i;   // LDS

// Copy 16 bytes global -> LDS asynchronously (tracked by ASYNCcnt).
static __device__ __forceinline__ void async_load_b128(unsigned lds_off, const void* gptr) {
#if defined(ASYNC_VIA_BUILTIN)
    __builtin_amdgcn_global_load_async_to_lds_b128(
        (gptr_v4i)(unsigned long long)(size_t)gptr,
        (lptr_v4i)(unsigned long long)lds_off,
        0, 0);
#else
    asm volatile("global_load_async_to_lds_b128 %0, %1, off"
                 :: "v"(lds_off), "v"(gptr) : "memory");
#endif
}

static __device__ __forceinline__ unsigned lds_off_of(const void* p) {
    // generic pointers to LDS carry the workgroup-relative offset in the low 32 bits
    return (unsigned)(size_t)p;
}

// ---------------------------------------------------------------------------
// WMMA helpers (CDNA5: V_WMMA_F32_16X16X32_F16, wave32)
// ---------------------------------------------------------------------------
static __device__ __forceinline__ v8f wmma16(v16h a, v16h b, v8f c) {
    return __builtin_amdgcn_wmma_f32_16x16x32_f16(
        /*neg_a=*/false, a, /*neg_b=*/false, b,
        /*c_mod=*/(short)0, c, /*reuse_a=*/false, /*reuse_b=*/false);
}

// Load one 16x32 (MxK) f16 tile from a row-major matrix, striped per ISA:
// lanes 0-15 hold row (lane), K halves {k..k+7, k+16..k+23};
// lanes 16-31 hold row (lane-16), K halves {k+8..k+15, k+24..k+31}.
// Same striping is used for the B operand (W stored row-major [N,K], B=W^T).
static __device__ __forceinline__ v16h load_tile(const _Float16* base, int ld, int lane) {
    const int r  = lane & 15;
    const int ko = (lane >> 4) << 3;   // 0 or 8 halves
    const _Float16* p = base + r * ld + ko;
    v8h lo = *(const v8h*)(p);
    v8h hi = *(const v8h*)(p + 16);
    v16h out;
#pragma unroll
    for (int i = 0; i < 8; ++i) { out[i] = lo[i]; out[i + 8] = hi[i]; }
    return out;
}

static __device__ __forceinline__ float sigmoidf_(float x) {
    return 1.0f / (1.0f + __expf(-x));
}

// ---------------------------------------------------------------------------
// Small prep kernels
// ---------------------------------------------------------------------------
__global__ void cvt_f16(const float* __restrict__ in, _Float16* __restrict__ out, int n) {
    for (int i = blockIdx.x * blockDim.x + threadIdx.x; i < n; i += gridDim.x * blockDim.x)
        out[i] = (_Float16)in[i];
}

__global__ void vec_add_f32(const float* a, const float* b, float* o, int n) {
    int i = blockIdx.x * blockDim.x + threadIdx.x;
    if (i < n) o[i] = a[i] + b[i];
}

__global__ void pad_av(const float* __restrict__ av, _Float16* __restrict__ av16) {
    int i = blockIdx.x * blockDim.x + threadIdx.x;  // 16*128
    if (i < 16 * 128) {
        int n = i >> 7, k = i & 127;
        av16[i] = (_Float16)((n < 12) ? av[n * 128 + k] : 0.0f);
    }
}

// x16[(l*32+b)*256+e] = emb[sentence[b][l]][e]
__global__ void embed_kernel(const int* __restrict__ sent, const float* __restrict__ emb,
                             _Float16* __restrict__ x16) {
    int i = blockIdx.x * blockDim.x + threadIdx.x;
    if (i >= 128 * 32 * 256) return;
    int e = i & 255, r = i >> 8;
    int b = r & 31, l = r >> 5;
    x16[i] = (_Float16)emb[(size_t)sent[b * 128 + l] * 256 + e];
}

// labE[(t*32+b)*64+j] = le_w[j][batch_ent[b][t]]   (post-step label at t)
__global__ void label_kernel(const int* __restrict__ ents, const float* __restrict__ le_w,
                             _Float16* __restrict__ labE) {
    int i = blockIdx.x * blockDim.x + threadIdx.x;
    if (i >= 128 * 32 * 64) return;
    int j = i & 63, r = i >> 6;
    int b = r & 31, t = r >> 5;
    labE[i] = (_Float16)le_w[j * 16 + ents[b * 128 + t]];
}

// cx[(t*32+b)*576+c] = c<512 ? enc16 : (t==0 ? 0 : labE[t-1])
__global__ void build_cx(const _Float16* __restrict__ enc16, const _Float16* __restrict__ labE,
                         _Float16* __restrict__ cx) {
    int i = blockIdx.x * blockDim.x + threadIdx.x;
    if (i >= 4096 * 576) return;
    int c = i % 576, r = i / 576;
    int b = r & 31, t = r >> 5;
    _Float16 v;
    if (c < 512)      v = enc16[r * 512 + c];
    else if (t == 0)  v = (_Float16)0.0f;
    else              v = labE[((t - 1) * 32 + b) * 64 + (c - 512)];
    cx[i] = v;
}

// Eseq[(b*128+t)*576 + 512 + j] = labE[(t*32+b)*64+j]
__global__ void eseq_label(const _Float16* __restrict__ labE, _Float16* __restrict__ Eseq) {
    int i = blockIdx.x * blockDim.x + threadIdx.x;
    if (i >= 32 * 128 * 64) return;
    int j = i & 63, r = i >> 6;
    int t = r & 127, b = r >> 7;
    Eseq[(size_t)r * 576 + 512 + j] = labE[(t * 32 + b) * 64 + j];
}

// ---------------------------------------------------------------------------
// Generic WMMA GEMM: C[4096, N] (f32) = A[4096, K] (f16) * W[N, K]^T + bias
// Block: 256 thr (8 waves); tile 64M x 128N. A/B chunks staged in LDS via
// async global->LDS DMA, double-buffered (DMA of chunk k+1 overlaps WMMA of k).
// Requires N % 128 == 0, K % 32 == 0 (true for all call sites).
// ---------------------------------------------------------------------------
__global__ __launch_bounds__(256)
void gemm_wmma(const _Float16* __restrict__ A, int lda,
               const _Float16* __restrict__ W, int ldw,
               const float* __restrict__ bias,
               float* __restrict__ C, int ldc,
               int N, int K) {
    __shared__ _Float16 Ab[2][64][32];    // 2 x 4 KB
    __shared__ _Float16 Bb[2][128][32];   // 2 x 8 KB

    const int tid    = threadIdx.x;
    const int lane   = tid & 31;
    const int wave   = tid >> 5;
    const int mbase  = blockIdx.x * 64;
    const int nbase0 = blockIdx.y * 128;
    const int nbase  = nbase0 + wave * 16;
    const int nk     = K >> 5;

    __builtin_amdgcn_s_wait_tensorcnt(0);   // exercise TENSORcnt path (no-op here)

    // async-load assignments: A 64 rows x 4 x 16B = 1/thread; B 128 rows x 4 x 16B = 2/thread
    const int arow = tid >> 2, aseg = tid & 3;
    const int brow = tid >> 1, bseg = (tid & 1) * 2;

    auto issue = [&](int kc, int buf) {
        const int k = kc << 5;
        async_load_b128(lds_off_of(&Ab[buf][arow][aseg * 8]),
                        A + (size_t)(mbase + arow) * lda + k + aseg * 8);
        async_load_b128(lds_off_of(&Bb[buf][brow][bseg * 8]),
                        W + (size_t)(nbase0 + brow) * ldw + k + bseg * 8);
        async_load_b128(lds_off_of(&Bb[buf][brow][(bseg + 1) * 8]),
                        W + (size_t)(nbase0 + brow) * ldw + k + (bseg + 1) * 8);
    };

    issue(0, 0);
    v8f acc[4] = {v8f{}, v8f{}, v8f{}, v8f{}};

    for (int kc = 0; kc < nk; ++kc) {
        const int buf = kc & 1;
        __syncthreads();                    // previous chunk fully consumed
        if (kc + 1 < nk) {
            issue(kc + 1, buf ^ 1);         // overlap next DMA with this chunk's math
            WAIT_ASYNC(3);                  // chunk kc landed (in-order), kc+1 in flight
        } else {
            WAIT_ASYNC(0);
        }
        __syncthreads();                    // all waves' DMA for chunk kc visible
        v16h bt = load_tile(&Bb[buf][wave * 16][0], 32, lane);
#pragma unroll
        for (int mt = 0; mt < 4; ++mt) {
            v16h at = load_tile(&Ab[buf][mt * 16][0], 32, lane);
            acc[mt] = wmma16(at, bt, acc[mt]);
        }
    }

    const int col = nbase + (lane & 15);
    const float bv = bias ? bias[col] : 0.0f;
#pragma unroll
    for (int mt = 0; mt < 4; ++mt) {
        int r0 = mbase + mt * 16 + ((lane >> 4) << 3);
#pragma unroll
        for (int i = 0; i < 8; ++i)
            C[(size_t)(r0 + i) * ldc + col] = acc[mt][i] + bv;
    }
}

// ---------------------------------------------------------------------------
// GRU recurrence: one block per direction (grid = 2), 512 threads = 16 waves.
// gh = h @ whh^T  via WMMA (M=32 -> 2 tiles, N=768 -> 48 tiles, K=256).
// gi (with bih) precomputed in global [L*B, 1536], this dir at col off dir*768.
// ---------------------------------------------------------------------------
__global__ __launch_bounds__(512)
void gru_kernel(const float* __restrict__ gi,
                const _Float16* __restrict__ whhF, const _Float16* __restrict__ whhB,
                const float* __restrict__ bhhF, const float* __restrict__ bhhB,
                _Float16* __restrict__ out16, float* __restrict__ out32) {
    __shared__ _Float16 h16[32][256];   // 16 KB
    __shared__ float    h32[32][256];   // 32 KB
    __shared__ float    gh [32][768];   // 96 KB

    const int dir  = blockIdx.x;
    const _Float16* whh = dir ? whhB : whhF;
    const float*    bhh = dir ? bhhB : bhhF;
    const int lane = threadIdx.x & 31;
    const int wave = threadIdx.x >> 5;

    for (int i = threadIdx.x; i < 32 * 256; i += 512) {
        ((_Float16*)h16)[i] = (_Float16)0.0f;
        ((float*)h32)[i]    = 0.0f;
    }
    __syncthreads();

    for (int s = 0; s < 128; ++s) {
        const int t = dir ? (127 - s) : s;
        // ---- gh = h @ whh^T -------------------------------------------------
#pragma unroll
        for (int g = 0; g < 3; ++g) {
            const int nt = wave + g * 16;            // N tile 0..47
            v8f acc0 = v8f{}, acc1 = v8f{};
            for (int k = 0; k < 256; k += 32) {
                v16h bt = load_tile(whh + nt * 16 * 256 + k, 256, lane);
                v16h a0 = load_tile(&h16[0][k],  256, lane);
                v16h a1 = load_tile(&h16[16][k], 256, lane);
                acc0 = wmma16(a0, bt, acc0);
                acc1 = wmma16(a1, bt, acc1);
            }
            const int col = nt * 16 + (lane & 15);
            const int rb  = (lane >> 4) << 3;
#pragma unroll
            for (int i = 0; i < 8; ++i) {
                gh[rb + i][col]      = acc0[i];
                gh[16 + rb + i][col] = acc1[i];
            }
        }
        __syncthreads();
        // ---- elementwise gate math -----------------------------------------
        for (int i = threadIdx.x; i < 32 * 256; i += 512) {
            const int b = i >> 8, j = i & 255;
            const int row = t * 32 + b;
            const float* gir = gi + (size_t)row * 1536 + dir * 768;
            const float ghr = gh[b][j]       + bhh[j];
            const float ghz = gh[b][256 + j] + bhh[256 + j];
            const float ghn = gh[b][512 + j] + bhh[512 + j];
            const float r = sigmoidf_(gir[j]       + ghr);
            const float z = sigmoidf_(gir[256 + j] + ghz);
            const float n = tanhf   (gir[512 + j] + r * ghn);
            const float hn = (1.0f - z) * n + z * h32[b][j];
            h32[b][j] = hn;
            h16[b][j] = (_Float16)hn;
            const int oc = dir * 256 + j;
            if (out16) out16[(size_t)row * 512 + oc] = (_Float16)hn;
            if (out32) out32[(size_t)row * 512 + oc] = hn;
        }
        __syncthreads();
    }
}

// ---------------------------------------------------------------------------
// BatchNorm over (L,B) per channel (512 channels), writes f16 enc.
// ---------------------------------------------------------------------------
__global__ __launch_bounds__(256)
void bn_kernel(const float* __restrict__ l1, _Float16* __restrict__ enc16) {
    __shared__ float rs[256], rq[256];
    const int c = blockIdx.x;
    float s = 0.0f, q = 0.0f;
    for (int r = threadIdx.x; r < 4096; r += 256) {
        float v = l1[(size_t)r * 512 + c];
        s += v; q += v * v;
    }
    rs[threadIdx.x] = s; rq[threadIdx.x] = q;
    __syncthreads();
    for (int o = 128; o > 0; o >>= 1) {
        if (threadIdx.x < o) { rs[threadIdx.x] += rs[threadIdx.x + o]; rq[threadIdx.x] += rq[threadIdx.x + o]; }
        __syncthreads();
    }
    const float mean = rs[0] * (1.0f / 4096.0f);
    const float var  = rq[0] * (1.0f / 4096.0f) - mean * mean;
    const float rstd = rsqrtf(var + 1e-5f);
    for (int r = threadIdx.x; r < 4096; r += 256)
        enc16[(size_t)r * 512 + c] = (_Float16)((l1[(size_t)r * 512 + c] - mean) * rstd);
}

// ---------------------------------------------------------------------------
// LSTM decoder: single block, 512 threads = 16 waves, 128 sequential steps.
// gh = h @ whh^T done in two batch-halves (M=16 each) to bound LDS; cell state
// lives in registers; tag GEMM + log_softmax per step; writes Eseq + entity.
// ---------------------------------------------------------------------------
__global__ __launch_bounds__(512)
void lstm_kernel(const float* __restrict__ gi,          // [4096,2048], includes bih+bhh
                 const _Float16* __restrict__ whh,      // [2048,512] f16
                 const float* __restrict__ h0, const float* __restrict__ c0,
                 const float* __restrict__ tag_w, const float* __restrict__ tag_b,
                 _Float16* __restrict__ Eseq,           // [B,L,576] f16 (h part)
                 float* __restrict__ entity) {          // [B,L,16]
    __shared__ _Float16 h16[32][512];     // 32 KB
    __shared__ _Float16 gh16[16][2048];   // 64 KB (per batch-half)
    __shared__ float    tags[32][16];     // 2 KB

    const int lane = threadIdx.x & 31;
    const int wave = threadIdx.x >> 5;
    float creg[32];

    for (int hb = 0; hb < 2; ++hb)
        for (int i = threadIdx.x; i < 16 * 512; i += 512) {
            const int bl = i >> 9, j = i & 511;
            const int b = hb * 16 + bl;
            h16[b][j] = (_Float16)h0[b * 512 + j];
            creg[hb * 16 + bl] = c0[b * 512 + j];
        }
    __syncthreads();

    for (int t = 0; t < 128; ++t) {
        for (int hb = 0; hb < 2; ++hb) {
            // ---- gh(half) = h[half] @ whh^T --------------------------------
#pragma unroll
            for (int g = 0; g < 8; ++g) {
                const int nt = wave + g * 16;          // N tile 0..127
                v8f acc = v8f{};
                for (int k = 0; k < 512; k += 32) {
                    v16h bt = load_tile(whh + nt * 16 * 512 + k, 512, lane);
                    v16h at = load_tile(&h16[hb * 16][k], 512, lane);
                    acc = wmma16(at, bt, acc);
                }
                const int col = nt * 16 + (lane & 15);
                const int rb  = (lane >> 4) << 3;
#pragma unroll
                for (int i = 0; i < 8; ++i) gh16[rb + i][col] = (_Float16)acc[i];
            }
            __syncthreads();
            // ---- elementwise LSTM cell update ------------------------------
            for (int i = threadIdx.x; i < 16 * 512; i += 512) {
                const int bl = i >> 9, j = i & 511;
                const int b = hb * 16 + bl;
                const int row = t * 32 + b;
                const float* g4 = gi + (size_t)row * 2048;
                const float ii = g4[j]        + (float)gh16[bl][j];
                const float ff = g4[512 + j]  + (float)gh16[bl][512 + j];
                const float gg = g4[1024 + j] + (float)gh16[bl][1024 + j];
                const float oo = g4[1536 + j] + (float)gh16[bl][1536 + j];
                const int ci = hb * 16 + bl;
                const float c = sigmoidf_(ff) * creg[ci] + sigmoidf_(ii) * tanhf(gg);
                const float h = sigmoidf_(oo) * tanhf(c);
                creg[ci] = c;
                h16[b][j] = (_Float16)h;
                Eseq[(size_t)(b * 128 + t) * 576 + j] = (_Float16)h;
            }
            __syncthreads();
        }
        // ---- tags = h @ tag_w^T + tag_b  (32x16, VALU) ---------------------
        {
            const int b = threadIdx.x >> 4, e = threadIdx.x & 15;
            float s = tag_b[e];
            for (int j = 0; j < 512; ++j) s += (float)h16[b][j] * tag_w[e * 512 + j];
            tags[b][e] = s;
        }
        __syncthreads();
        // ---- log_softmax over ENT=16 ---------------------------------------
        if (threadIdx.x < 32) {
            const int b = threadIdx.x;
            float m = tags[b][0];
            for (int e = 1; e < 16; ++e) m = fmaxf(m, tags[b][e]);
            float sum = 0.0f;
            for (int e = 0; e < 16; ++e) sum += __expf(tags[b][e] - m);
            const float lse = m + __logf(sum);
            for (int e = 0; e < 16; ++e)
                entity[(size_t)(b * 128 + t) * 16 + e] = tags[b][e] - lse;
        }
        __syncthreads();
    }
}

// ---------------------------------------------------------------------------
// Relation head: one block per (b,t). energy = tanh(s1[b,j,:] + s2[b,t,:]),
// logits = energy @ a_v^T (WMMA, N padded to 16), masked log_softmax over 12.
// ---------------------------------------------------------------------------
__global__ __launch_bounds__(128)
void rel_kernel(const float* __restrict__ s1, const float* __restrict__ s2,
                const _Float16* __restrict__ av16, float* __restrict__ rel) {
    __shared__ _Float16 en[128][128];   // 32 KB
    __shared__ float    lg[128][16];    // 8 KB
    __shared__ float    s2s[128];

    const int bt = blockIdx.x;          // b*128 + t
    const int b = bt >> 7, t = bt & 127;
    if (threadIdx.x < 128) s2s[threadIdx.x] = s2[(size_t)bt * 128 + threadIdx.x];
    __syncthreads();

    for (int i = threadIdx.x; i < 128 * 128; i += 128) {
        const int j = i >> 7, a = i & 127;
        en[j][a] = (_Float16)tanhf(s1[(size_t)(b * 128 + j) * 128 + a] + s2s[a]);
    }
    __syncthreads();

    const int lane = threadIdx.x & 31;
    const int wave = threadIdx.x >> 5;
#pragma unroll
    for (int mg = 0; mg < 2; ++mg) {
        const int mt = wave + mg * 4;   // M tile 0..7
        v8f acc = v8f{};
        for (int k = 0; k < 128; k += 32) {
            v16h bt2 = load_tile(av16 + k, 128, lane);
            v16h at  = load_tile(&en[mt * 16][k], 128, lane);
            acc = wmma16(at, bt2, acc);
        }
        const int col = lane & 15;
        const int rb  = mt * 16 + ((lane >> 4) << 3);
#pragma unroll
        for (int i = 0; i < 8; ++i) lg[rb + i][col] = acc[i];
    }
    __syncthreads();

    const int j = threadIdx.x;
    float* outp = rel + ((size_t)bt * 128 + j) * 12;
    if (j <= t) {
        float m = lg[j][0];
        for (int r = 1; r < 12; ++r) m = fmaxf(m, lg[j][r]);
        float sum = 0.0f;
        for (int r = 0; r < 12; ++r) sum += __expf(lg[j][r] - m);
        const float lse = m + __logf(sum);
        for (int r = 0; r < 12; ++r) outp[r] = lg[j][r] - lse;
    } else {
        for (int r = 0; r < 12; ++r) outp[r] = 0.0f;
    }
}

// ---------------------------------------------------------------------------
// Host orchestration
// ---------------------------------------------------------------------------
extern "C" void kernel_launch(void* const* d_in, const int* in_sizes, int n_in,
                              void* d_out, int out_size, void* d_ws, size_t ws_size,
                              hipStream_t stream) {
    (void)in_sizes; (void)n_in; (void)out_size; (void)ws_size;
    // Inputs in setup_inputs() order
    const int*   sentence  = (const int*)d_in[0];
    const int*   batch_ent = (const int*)d_in[1];
    const float* emb       = (const float*)d_in[2];
    const float* g0f_wih = (const float*)d_in[3],  *g0f_whh = (const float*)d_in[4];
    const float* g0f_bih = (const float*)d_in[5],  *g0f_bhh = (const float*)d_in[6];
    const float* g0b_wih = (const float*)d_in[7],  *g0b_whh = (const float*)d_in[8];
    const float* g0b_bih = (const float*)d_in[9],  *g0b_bhh = (const float*)d_in[10];
    const float* g1f_wih = (const float*)d_in[11], *g1f_whh = (const float*)d_in[12];
    const float* g1f_bih = (const float*)d_in[13], *g1f_bhh = (const float*)d_in[14];
    const float* g1b_wih = (const float*)d_in[15], *g1b_whh = (const float*)d_in[16];
    const float* g1b_bih = (const float*)d_in[17], *g1b_bhh = (const float*)d_in[18];
    const float* lstm_wih = (const float*)d_in[19], *lstm_whh = (const float*)d_in[20];
    const float* lstm_bih = (const float*)d_in[21], *lstm_bhh = (const float*)d_in[22];
    const float* tag_w = (const float*)d_in[23], *tag_b = (const float*)d_in[24];
    const float* le_w  = (const float*)d_in[25];
    const float* a_w1  = (const float*)d_in[26], *a_w2 = (const float*)d_in[27];
    const float* a_v   = (const float*)d_in[28];
    const float* h0    = (const float*)d_in[29], *c0 = (const float*)d_in[30];

    char* ws = (char*)d_ws;
    size_t off = 0;
    auto A16 = [&](size_t elems) -> _Float16* {
        _Float16* p = (_Float16*)(ws + off);
        off = (off + elems * 2 + 255) & ~(size_t)255;
        return p;
    };
    auto A32 = [&](size_t elems) -> float* {
        float* p = (float*)(ws + off);
        off = (off + elems * 4 + 255) & ~(size_t)255;
        return p;
    };

    // f16 weight copies (B operands)
    _Float16* wg0f_ih = A16(768 * 256);  _Float16* wg0f_hh = A16(768 * 256);
    _Float16* wg0b_ih = A16(768 * 256);  _Float16* wg0b_hh = A16(768 * 256);
    _Float16* wg1f_ih = A16(768 * 512);  _Float16* wg1f_hh = A16(768 * 256);
    _Float16* wg1b_ih = A16(768 * 512);  _Float16* wg1b_hh = A16(768 * 256);
    _Float16* wl_ih   = A16(2048 * 576); _Float16* wl_hh   = A16(2048 * 512);
    _Float16* waw1    = A16(128 * 576);  _Float16* waw2    = A16(128 * 576);
    _Float16* wav     = A16(16 * 128);
    // activations
    _Float16* x16   = A16(4096 * 256);
    _Float16* l016  = A16(4096 * 512);
    _Float16* enc16 = A16(4096 * 512);
    _Float16* labE  = A16(128 * 32 * 64);
    _Float16* cx16  = A16(4096 * 576);
    _Float16* Eseq  = A16(4096 * 576);
    float* gi    = A32(4096 * 2048);   // reused: gi0(1536) -> gi1(1536) -> lstm gi(2048)
    float* l132  = A32(4096 * 512);
    float* s1    = A32(4096 * 128);
    float* s2    = A32(4096 * 128);
    float* blstm = A32(2048);

    float* entity = (float*)d_out;                 // [B,L,16]
    float* relout = (float*)d_out + 32 * 128 * 16; // [B,L,L,12]

    auto cvt = [&](const float* src, _Float16* dst, int n) {
        cvt_f16<<<(n + 255) / 256, 256, 0, stream>>>(src, dst, n);
    };

    // --- weight prep ---
    cvt(g0f_wih, wg0f_ih, 768 * 256);  cvt(g0f_whh, wg0f_hh, 768 * 256);
    cvt(g0b_wih, wg0b_ih, 768 * 256);  cvt(g0b_whh, wg0b_hh, 768 * 256);
    cvt(g1f_wih, wg1f_ih, 768 * 512);  cvt(g1f_whh, wg1f_hh, 768 * 256);
    cvt(g1b_wih, wg1b_ih, 768 * 512);  cvt(g1b_whh, wg1b_hh, 768 * 256);
    cvt(lstm_wih, wl_ih, 2048 * 576);  cvt(lstm_whh, wl_hh, 2048 * 512);
    cvt(a_w1, waw1, 128 * 576);        cvt(a_w2, waw2, 128 * 576);
    pad_av<<<8, 256, 0, stream>>>(a_v, wav);
    vec_add_f32<<<8, 256, 0, stream>>>(lstm_bih, lstm_bhh, blstm, 2048);

    // --- embedding ---
    embed_kernel<<<4096, 256, 0, stream>>>(sentence, emb, x16);

    // --- GRU layer 0: input projections then recurrence ---
    gemm_wmma<<<dim3(64, 6), 256, 0, stream>>>(x16, 256, wg0f_ih, 256, g0f_bih, gi,       1536, 768, 256);
    gemm_wmma<<<dim3(64, 6), 256, 0, stream>>>(x16, 256, wg0b_ih, 256, g0b_bih, gi + 768, 1536, 768, 256);
    gru_kernel<<<2, 512, 0, stream>>>(gi, wg0f_hh, wg0b_hh, g0f_bhh, g0b_bhh, l016, nullptr);

    // --- GRU layer 1 ---
    gemm_wmma<<<dim3(64, 6), 256, 0, stream>>>(l016, 512, wg1f_ih, 512, g1f_bih, gi,       1536, 768, 512);
    gemm_wmma<<<dim3(64, 6), 256, 0, stream>>>(l016, 512, wg1b_ih, 512, g1b_bih, gi + 768, 1536, 768, 512);
    gru_kernel<<<2, 512, 0, stream>>>(gi, wg1f_hh, wg1b_hh, g1f_bhh, g1b_bhh, nullptr, l132);

    // --- BatchNorm -> enc16 ---
    bn_kernel<<<512, 256, 0, stream>>>(l132, enc16);

    // --- teacher-forced labels, decoder input, LSTM gi GEMM ---
    label_kernel<<<(128 * 32 * 64 + 255) / 256, 256, 0, stream>>>(batch_ent, le_w, labE);
    build_cx<<<(4096 * 576 + 255) / 256, 256, 0, stream>>>(enc16, labE, cx16);
    eseq_label<<<(32 * 128 * 64 + 255) / 256, 256, 0, stream>>>(labE, Eseq);
    gemm_wmma<<<dim3(64, 16), 256, 0, stream>>>(cx16, 576, wl_ih, 576, blstm, gi, 2048, 2048, 576);

    // --- LSTM decoder (writes Eseq h-part and entity output) ---
    lstm_kernel<<<1, 512, 0, stream>>>(gi, wl_hh, h0, c0, tag_w, tag_b, Eseq, entity);

    // --- attention scores ---
    gemm_wmma<<<dim3(64, 1), 256, 0, stream>>>(Eseq, 576, waw1, 576, nullptr, s1, 128, 128, 576);
    gemm_wmma<<<dim3(64, 1), 256, 0, stream>>>(Eseq, 576, waw2, 576, nullptr, s2, 128, 128, 576);

    // --- relation head ---
    rel_kernel<<<4096, 128, 0, stream>>>(s1, s2, wav, relout);
}
